// Aggregator_16707422781624
// MI455X (gfx1250) — matrix operand order repaired and checked by reading, not compiled, not run on hardware
//
#include <hip/hip_runtime.h>

typedef __attribute__((ext_vector_type(16))) __bf16 v16bf;
typedef __attribute__((ext_vector_type(8)))  __bf16 v8bf;
typedef __attribute__((ext_vector_type(8)))  float  v8f;

#define DIMS 128   // D
#define KNBR 32    // neighbors per node

// ---------------------------------------------------------------------------
// Prep: transpose + convert W[d][e] (f32, row-major) -> Wt[e][d] (bf16).
// 128x128 = 16384 elements = 32 KB bf16 in workspace; L2-resident afterwards.
// ---------------------------------------------------------------------------
__global__ __launch_bounds__(256) void wt_prep_kernel(
    const float* __restrict__ W, unsigned short* __restrict__ WtRaw)
{
    int t = blockIdx.x * blockDim.x + threadIdx.x;   // 0..16383
    int e = t >> 7;          // output row    (= W column)
    int d = t & 127;         // output column (= W row)
    __bf16* Wt = (__bf16*)WtRaw;
    Wt[t] = (__bf16)W[d * DIMS + e];
}

// ---------------------------------------------------------------------------
// Main: one block (8 waves) per target node.
//   LDS:  Xs[32][128] bf16 = gathered neighbor rows (8 KB)
//   wave w -> N-tile columns [16w, 16w+16)
//   acc0 = rows 0..15, acc1 = rows 16..31, K swept in 4 steps of 32
// ---------------------------------------------------------------------------
__global__ __launch_bounds__(256) void pool_agg_kernel(
    const float*          __restrict__ feat,   // [V,128] f32
    const float*          __restrict__ bias,   // [128]   f32
    const int*            __restrict__ nidx,   // [B,32]
    const unsigned short* __restrict__ WtRaw,  // [128,128] bf16 (e-major)
    float*                __restrict__ out)    // [B,128] f32
{
    __shared__ __bf16 Xs[KNBR * DIMS];         // 8 KB
    const __bf16* Wt = (const __bf16*)WtRaw;

    const int blk = blockIdx.x;
    const int tid = threadIdx.x;

    // ---- gather 32 neighbor rows, cvt f32->bf16 into LDS -------------------
    {
        const int r   = tid >> 3;              // row 0..31
        const int seg = tid & 7;               // 16-float segment
        const int nid = nidx[blk * KNBR + r];
        const float4* src = (const float4*)(feat + (long)nid * DIMS + seg * 16);
        float4 f0 = src[0], f1 = src[1], f2 = src[2], f3 = src[3];
        v8bf a, b;
        a[0]=(__bf16)f0.x; a[1]=(__bf16)f0.y; a[2]=(__bf16)f0.z; a[3]=(__bf16)f0.w;
        a[4]=(__bf16)f1.x; a[5]=(__bf16)f1.y; a[6]=(__bf16)f1.z; a[7]=(__bf16)f1.w;
        b[0]=(__bf16)f2.x; b[1]=(__bf16)f2.y; b[2]=(__bf16)f2.z; b[3]=(__bf16)f2.w;
        b[4]=(__bf16)f3.x; b[5]=(__bf16)f3.y; b[6]=(__bf16)f3.z; b[7]=(__bf16)f3.w;
        v8bf* dst = (v8bf*)&Xs[r * DIMS + seg * 16];
        dst[0] = a;
        dst[1] = b;
    }
    __syncthreads();

    // ---- WMMA: Z[32x128] = X[32x128] @ W[128x128] -------------------------
    const int lane = tid & 31;
    const int wave = tid >> 5;                 // N-tile id 0..7
    const int col  = lane & 15;
    const int half = lane >> 4;                // 0: lanes 0-15, 1: lanes 16-31
    const int nc   = wave * 16 + col;          // output column for this lane

    v8f acc0 = {};                             // rows 0..15
    v8f acc1 = {};                             // rows 16..31

    const __bf16* wrow  = Wt + nc * DIMS;                // B: column nc of W
    const __bf16* arow0 = &Xs[col * DIMS];               // A tile 0 (rows 0-15)
    const __bf16* arow1 = &Xs[(col + 16) * DIMS];        // A tile 1 (rows 16-31)

#pragma unroll
    for (int kb = 0; kb < 4; ++kb) {
        const int kbase = kb * 32;

        // B fragment: lane half selects k runs {0..7,16..23} vs {8..15,24..31}
        const int ks = kbase + half * 8;
        v8bf blo = *(const v8bf*)(wrow + ks);
        v8bf bhi = *(const v8bf*)(wrow + ks + 16);
        v16bf Bf = __builtin_shufflevector(blo, bhi,
                       0,1,2,3,4,5,6,7,8,9,10,11,12,13,14,15);

        // A fragments: per ISA layout, lane half selects k runs {0..7,16..23}
        // vs {8..15,24..31} within the 32-wide K block.
        const int ka = kbase + half * 8;
        v8bf a0lo = *(const v8bf*)(arow0 + ka);
        v8bf a0hi = *(const v8bf*)(arow0 + ka + 16);
        v8bf a1lo = *(const v8bf*)(arow1 + ka);
        v8bf a1hi = *(const v8bf*)(arow1 + ka + 16);
        v16bf A0 = __builtin_shufflevector(a0lo, a0hi,
                       0,1,2,3,4,5,6,7,8,9,10,11,12,13,14,15);
        v16bf A1 = __builtin_shufflevector(a1lo, a1hi,
                       0,1,2,3,4,5,6,7,8,9,10,11,12,13,14,15);

        acc0 = __builtin_amdgcn_wmma_f32_16x16x32_bf16(
                   false, A0, false, Bf, (short)0, acc0, false, false);
        acc1 = __builtin_amdgcn_wmma_f32_16x16x32_bf16(
                   false, A1, false, Bf, (short)0, acc1, false, false);
    }

    // ---- max over 32 neighbors, bias, sigmoid -----------------------------
    // Per C/D layout each lane holds M = {i or 8+i} per VGPR i for its half;
    // acc0+acc1 cover 16 of the 32 rows per lane; xor-16 shuffle merges halves.
    float m = acc0[0];
#pragma unroll
    for (int i = 1; i < 8; ++i) m = fmaxf(m, acc0[i]);
#pragma unroll
    for (int i = 0; i < 8; ++i) m = fmaxf(m, acc1[i]);
    m = fmaxf(m, __shfl_xor(m, 16, 32));

    float z = m + bias[nc];
    float h = 1.0f / (1.0f + __expf(-z));
    if (lane < 16)
        out[(long)blk * DIMS + nc] = h;
}

// ---------------------------------------------------------------------------
extern "C" void kernel_launch(void* const* d_in, const int* in_sizes, int n_in,
                              void* d_out, int out_size, void* d_ws, size_t ws_size,
                              hipStream_t stream)
{
    const float* feat = (const float*)d_in[0];   // [V,128]
    const float* W    = (const float*)d_in[1];   // [128,128]
    const float* bias = (const float*)d_in[2];   // [128]
    const int*   idx  = (const int*)d_in[3];     // [B,32]
    float*       out  = (float*)d_out;
    unsigned short* Wt = (unsigned short*)d_ws;  // needs 32 KB of workspace

    const int B = in_sizes[3] / KNBR;            // 50,000

    wt_prep_kernel<<<(DIMS * DIMS) / 256, 256, 0, stream>>>(W, Wt);
    pool_agg_kernel<<<B, 256, 0, stream>>>(feat, bias, idx, Wt, out);
}
